// EncoderTransformer_46677704573536
// MI455X (gfx1250) — matrix operand order repaired and checked by reading, not compiled, not run on hardware
//
#include <hip/hip_runtime.h>

// ---------- types & helpers ----------
typedef __attribute__((ext_vector_type(16))) __bf16 v16bf;
typedef __attribute__((ext_vector_type(8)))  float  v8f;

union FragU { v16bf v; unsigned int u[8]; };

__device__ __forceinline__ unsigned short f2bf(float f) {
    unsigned int u = __float_as_uint(f);
    unsigned int r = u + 0x7FFFu + ((u >> 16) & 1u);   // round-to-nearest-even
    return (unsigned short)(r >> 16);
}

__device__ __forceinline__ v8f wmma_bf16(v16bf a, v16bf b, v8f c) {
    return __builtin_amdgcn_wmma_f32_16x16x32_bf16(
        false, a, false, b, (short)0, c, false, false);
}

// A-fragment (16x32, M=lane&15): pair p = 8*(r>>2) + 4*(l>>4) + (r&3)
__device__ __forceinline__ v16bf load_a(const unsigned int* base, int rowStrideU32, int lane) {
    FragU f;
    const unsigned int* rp = base + (lane & 15) * rowStrideU32;
    int q = (lane >> 4) << 2;
#pragma unroll
    for (int r = 0; r < 8; ++r)
        f.u[r] = rp[((r >> 2) << 3) + q + (r & 3)];
    return f.v;
}

// B-fragment (32x16, N=lane&15): pair p = 8*(r>>2) + 4*(l>>4) + (r&3)
__device__ __forceinline__ v16bf load_b(const unsigned int* base, int rowStrideU32, int lane) {
    FragU f;
    const unsigned int* rp = base + (lane & 15) * rowStrideU32 + ((lane >> 4) << 2);
#pragma unroll
    for (int r = 0; r < 8; ++r)
        f.u[r] = rp[((r >> 2) << 3) + (r & 3)];
    return f.v;
}

// ---------- CDNA5 async global->LDS (ASYNCcnt) ----------
__device__ __forceinline__ unsigned lds_off(const void* p) {
    // generic shared pointer: low 32 bits are the wave-relative LDS byte offset
    return (unsigned)(unsigned long long)p;
}
__device__ __forceinline__ void async_b128(unsigned lds, const void* gptr) {
    asm volatile("global_load_async_to_lds_b128 %0, %1, off"
                 :: "v"(lds), "v"(gptr) : "memory");
}
__device__ __forceinline__ void wait_async0() {
    asm volatile("s_wait_asynccnt 0" ::: "memory");
}

// ---------- weight transpose+convert: WT[l][n][k] = bf16(W[l][k][n]) ----------
__global__ __launch_bounds__(256) void convert_w_kernel(
    const float* Wq, const float* Wk, const float* Wv, const float* W1, const float* W2,
    unsigned short* WqT, unsigned short* WkT, unsigned short* WvT,
    unsigned short* W1T, unsigned short* W2T)
{
    int which = blockIdx.y;
    const float* src = which == 0 ? Wq : which == 1 ? Wk : which == 2 ? Wv : which == 3 ? W1 : W2;
    unsigned short* dst = which == 0 ? WqT : which == 1 ? WkT : which == 2 ? WvT : which == 3 ? W1T : W2T;
    int l = blockIdx.x >> 8, n = blockIdx.x & 255, k = threadIdx.x;
    dst[(size_t)l * 65536 + (size_t)n * 256 + k] =
        f2bf(src[(size_t)l * 65536 + (size_t)k * 256 + n]);
}

// ---------- embed: H = zs @ W_in + b_in ----------
__global__ __launch_bounds__(256) void embed_kernel(
    const float* xs, const float* ys, const float* W_in, const float* b_in,
    float* H, unsigned short* Hb)
{
    int row = blockIdx.x;        // b*1024 + i
    int col = threadIdx.x;
    int i = row & 1023;
    float acc = b_in[col];
    const float* xr = xs + (size_t)row * 63;
    for (int d = 0; d < 63; ++d)
        acc += xr[d] * W_in[d * 256 + col];
    float yv = (i == 1023) ? 0.f : ys[row];
    acc += yv * W_in[63 * 256 + col];
    H[(size_t)row * 256 + col] = acc;
    Hb[(size_t)row * 256 + col] = f2bf(acc);
}

// ---------- generic 256-K GEMM, double-buffered async tiles ----------
// OUTMODE: 0 = bf16 [M][256], 1 = bf16 transposed per batch [B][256][1024], 2 = f32 [M][256]
template<bool BIAS, bool RELU, int OUTMODE>
__global__ __launch_bounds__(256) void gemm256_kernel(
    const unsigned int* __restrict__ A,      // bf16 pairs [M][128]
    const unsigned int* __restrict__ BT,     // bf16 pairs [256 n][128 k]
    const float* __restrict__ bias,
    float* __restrict__ outF,
    unsigned short* __restrict__ outB)
{
    __shared__ __align__(16) unsigned int As[2][2048];
    __shared__ __align__(16) unsigned int Bs[2][2048];

    const int tid = threadIdx.x, lane = tid & 31, wave = tid >> 5;
    const int m0 = blockIdx.x * 128, n0 = blockIdx.y * 128;

    auto fill = [&](int buf, int kb) {
#pragma unroll
        for (int t = 0; t < 2; ++t) {
            int e = tid + t * 256;                 // b128 units (512 per tile)
            int row = e >> 2, col = (e & 3) << 2;  // u32 row/col within tile
            async_b128(lds_off(&As[buf][e << 2]),
                       A + (size_t)(m0 + row) * 128 + kb * 16 + col);
            async_b128(lds_off(&Bs[buf][e << 2]),
                       BT + (size_t)(n0 + row) * 128 + kb * 16 + col);
        }
    };

    v8f acc[8];
#pragma unroll
    for (int i = 0; i < 8; ++i) acc[i] = (v8f)(0.f);

    fill(0, 0);
    for (int kb = 0; kb < 8; ++kb) {
        int cur = kb & 1;
        wait_async0();
        __syncthreads();
        if (kb < 7) fill(cur ^ 1, kb + 1);

        v16bf a = load_a(&As[cur][wave * 256], 16, lane);
#pragma unroll
        for (int nt = 0; nt < 8; ++nt) {
            v16bf b = load_b(&Bs[cur][nt * 256], 16, lane);
            acc[nt] = wmma_bf16(a, b, acc[nt]);
        }
    }

    const int mBase = m0 + wave * 16 + ((lane >> 4) << 3);
    const int nLane = lane & 15;
#pragma unroll
    for (int nt = 0; nt < 8; ++nt) {
        int n = n0 + nt * 16 + nLane;
        float bv = BIAS ? bias[n] : 0.f;
#pragma unroll
        for (int r = 0; r < 8; ++r) {
            int m = mBase + r;
            float v = acc[nt][r] + bv;
            if (RELU) v = fmaxf(v, 0.f);
            if (OUTMODE == 2)      outF[(size_t)m * 256 + n] = v;
            else if (OUTMODE == 0) outB[(size_t)m * 256 + n] = f2bf(v);
            else {
                int bb = m >> 10, ii = m & 1023;
                outB[((size_t)bb * 256 + n) * 1024 + ii] = f2bf(v);
            }
        }
    }
}

// ---------- fused attention: O = (relu(Q K^T)/n) @ V, double-buffered ----------
__global__ __launch_bounds__(256) void attn_kernel(
    const unsigned int* __restrict__ Q,   // bf16 pairs [B*N][128]
    const unsigned int* __restrict__ K,   // bf16 pairs [B*N][128]
    const unsigned int* __restrict__ VT,  // bf16 pairs [B][256][512]
    float* __restrict__ O)                // [B*N][256]
{
    __shared__ __align__(16) unsigned int Ks[2][4096];   // 32 j-rows x 256 d
    __shared__ __align__(16) unsigned int VTs[2][4096];  // 256 d-rows x 32 j

    const int tid = threadIdx.x, lane = tid & 31, wave = tid >> 5;
    const int b = blockIdx.y;
    const int m0 = b * 1024 + blockIdx.x * 128 + wave * 16;

    auto fill = [&](int buf, int j0) {
#pragma unroll
        for (int t = 0; t < 4; ++t) {
            int e = tid + t * 256;                   // b128 units (1024 per tile)
            int jj = e >> 5, ck = (e & 31) << 2;
            async_b128(lds_off(&Ks[buf][e << 2]),
                       K + (size_t)(b * 1024 + j0 + jj) * 128 + ck);
            int d = e >> 2, cv = (e & 3) << 2;
            async_b128(lds_off(&VTs[buf][e << 2]),
                       VT + ((size_t)b * 256 + d) * 512 + (j0 >> 1) + cv);
        }
    };

    // Q fragments for the wave's 16 rows, 8 k-steps (kept in VGPRs)
    v16bf qa[8];
    {
        const unsigned int* qrow = Q + (size_t)(m0 + (lane & 15)) * 128;
        int q4 = (lane >> 4) << 2;
#pragma unroll
        for (int kb = 0; kb < 8; ++kb) {
            FragU f;
#pragma unroll
            for (int r = 0; r < 8; ++r)
                f.u[r] = qrow[kb * 16 + ((r >> 2) << 3) + q4 + (r & 3)];
            qa[kb] = f.v;
        }
    }

    v8f acc[16];
#pragma unroll
    for (int i = 0; i < 16; ++i) acc[i] = (v8f)(0.f);

    const float inv_n = 1.0f / 1024.0f;

    fill(0, 0);
    for (int it = 0; it < 32; ++it) {
        int cur = it & 1;
        wait_async0();
        __syncthreads();
        if (it < 31) fill(cur ^ 1, (it + 1) * 32);

        // S = relu(Q K^T) * inv_n   (16 rows x 32 cols, kept in regs first)
        v8f s01[2];
#pragma unroll
        for (int jt = 0; jt < 2; ++jt) {
            v8f s = (v8f)(0.f);
#pragma unroll
            for (int kb = 0; kb < 8; ++kb) {
                v16bf kf = load_b(&Ks[cur][(jt * 16) * 128 + kb * 16], 128, lane);
                s = wmma_bf16(qa[kb], kf, s);
            }
            s01[jt] = s;
        }
        __syncthreads();   // everyone done reading Ks[cur] -> reuse as S scratch

        unsigned short* myS = (unsigned short*)&Ks[cur][0] + wave * 512;
        {
            int mb = (lane >> 4) << 3;
#pragma unroll
            for (int jt = 0; jt < 2; ++jt) {
                int jl = jt * 16 + (lane & 15);
#pragma unroll
                for (int r = 0; r < 8; ++r)
                    myS[(mb + r) * 32 + jl] = f2bf(fmaxf(s01[jt][r], 0.f) * inv_n);
            }
        }
        asm volatile("s_wait_dscnt 0" ::: "memory");

        // O += S @ V
        v16bf sa = load_a((const unsigned int*)myS, 16, lane);
#pragma unroll
        for (int dt = 0; dt < 16; ++dt) {
            v16bf vb = load_b(&VTs[cur][dt * 256], 16, lane);
            acc[dt] = wmma_bf16(sa, vb, acc[dt]);
        }
    }

    const int mBase = m0 + ((lane >> 4) << 3);
    const int nLane = lane & 15;
#pragma unroll
    for (int dt = 0; dt < 16; ++dt) {
        int n = dt * 16 + nLane;
#pragma unroll
        for (int r = 0; r < 8; ++r)
            O[(size_t)(mBase + r) * 256 + n] = acc[dt][r];
    }
}

// ---------- residual + LayerNorm (one wave32 per row) ----------
__global__ __launch_bounds__(256) void add_ln_kernel(
    float* H, const float* O, const float* g, const float* be, unsigned short* Hb)
{
    int row = blockIdx.x * 8 + (threadIdx.x >> 5);
    int lane = threadIdx.x & 31;
    float v[8]; float s = 0.f;
#pragma unroll
    for (int e = 0; e < 8; ++e) {
        int idx = lane + e * 32;
        v[e] = H[(size_t)row * 256 + idx] + O[(size_t)row * 256 + idx];
        s += v[e];
    }
#pragma unroll
    for (int off = 16; off > 0; off >>= 1) s += __shfl_xor(s, off, 32);
    float mu = s * (1.f / 256.f);
    float q = 0.f;
#pragma unroll
    for (int e = 0; e < 8; ++e) { float d = v[e] - mu; q += d * d; }
#pragma unroll
    for (int off = 16; off > 0; off >>= 1) q += __shfl_xor(q, off, 32);
    float rstd = rsqrtf(q * (1.f / 256.f) + 1e-5f);
#pragma unroll
    for (int e = 0; e < 8; ++e) {
        int idx = lane + e * 32;
        float o = (v[e] - mu) * rstd * g[idx] + be[idx];
        H[(size_t)row * 256 + idx] = o;
        Hb[(size_t)row * 256 + idx] = f2bf(o);
    }
}

// ---------- readout head ----------
__global__ __launch_bounds__(256) void head_kernel(
    const float* H, const float* W_out, const float* b_out, float* out)
{
    int row = blockIdx.x * 8 + (threadIdx.x >> 5);
    int lane = threadIdx.x & 31;
    float s = 0.f;
#pragma unroll
    for (int e = 0; e < 8; ++e) {
        int idx = lane + e * 32;
        s += H[(size_t)row * 256 + idx] * W_out[idx];
    }
#pragma unroll
    for (int off = 16; off > 0; off >>= 1) s += __shfl_xor(s, off, 32);
    if (lane == 0) out[row] = s + b_out[0];
}

// ---------- host orchestration ----------
extern "C" void kernel_launch(void* const* d_in, const int* in_sizes, int n_in,
                              void* d_out, int out_size, void* d_ws, size_t ws_size,
                              hipStream_t stream)
{
    const float* xs    = (const float*)d_in[0];
    const float* ys    = (const float*)d_in[1];
    const float* W_in  = (const float*)d_in[2];
    const float* b_in  = (const float*)d_in[3];
    const float* Wq    = (const float*)d_in[4];
    const float* Wk    = (const float*)d_in[5];
    const float* Wv    = (const float*)d_in[6];
    const float* g1    = (const float*)d_in[7];
    const float* be1   = (const float*)d_in[8];
    const float* W1    = (const float*)d_in[9];
    const float* b1    = (const float*)d_in[10];
    const float* W2    = (const float*)d_in[11];
    const float* b2    = (const float*)d_in[12];
    const float* g2    = (const float*)d_in[13];
    const float* be2   = (const float*)d_in[14];
    const float* W_out = (const float*)d_in[15];
    const float* b_out = (const float*)d_in[16];

    const int M = 32 * 1024;          // B*N rows
    char* ws = (char*)d_ws;
    size_t off = 0;
    auto alloc = [&](size_t bytes) {
        void* p = ws + off;
        off += (bytes + 255) & ~(size_t)255;
        return p;
    };
    float*          H   = (float*)         alloc((size_t)M * 256 * 4);
    unsigned short* Hb  = (unsigned short*)alloc((size_t)M * 256 * 2);
    unsigned short* Qb  = (unsigned short*)alloc((size_t)M * 256 * 2);   // reused as MLP hidden
    unsigned short* Kb  = (unsigned short*)alloc((size_t)M * 256 * 2);
    unsigned short* VTb = (unsigned short*)alloc((size_t)M * 256 * 2);   // [B][256][1024]
    float*          O   = (float*)         alloc((size_t)M * 256 * 4);
    unsigned short* WqT = (unsigned short*)alloc((size_t)12 * 65536 * 2);
    unsigned short* WkT = (unsigned short*)alloc((size_t)12 * 65536 * 2);
    unsigned short* WvT = (unsigned short*)alloc((size_t)12 * 65536 * 2);
    unsigned short* W1T = (unsigned short*)alloc((size_t)12 * 65536 * 2);
    unsigned short* W2T = (unsigned short*)alloc((size_t)12 * 65536 * 2);
    (void)ws_size; (void)in_sizes; (void)n_in; (void)out_size;

    convert_w_kernel<<<dim3(12 * 256, 5), 256, 0, stream>>>(
        Wq, Wk, Wv, W1, W2, WqT, WkT, WvT, W1T, W2T);

    embed_kernel<<<dim3(M), 256, 0, stream>>>(xs, ys, W_in, b_in, H, Hb);

    const unsigned int* Hbu = (const unsigned int*)Hb;
    for (int l = 0; l < 12; ++l) {
        gemm256_kernel<false, false, 0><<<dim3(M / 128, 2), 256, 0, stream>>>(
            Hbu, (const unsigned int*)(WqT + (size_t)l * 65536), nullptr, nullptr, Qb);
        gemm256_kernel<false, false, 0><<<dim3(M / 128, 2), 256, 0, stream>>>(
            Hbu, (const unsigned int*)(WkT + (size_t)l * 65536), nullptr, nullptr, Kb);
        gemm256_kernel<false, false, 1><<<dim3(M / 128, 2), 256, 0, stream>>>(
            Hbu, (const unsigned int*)(WvT + (size_t)l * 65536), nullptr, nullptr, VTb);

        attn_kernel<<<dim3(8, 32), 256, 0, stream>>>(
            (const unsigned int*)Qb, (const unsigned int*)Kb,
            (const unsigned int*)VTb, O);

        add_ln_kernel<<<dim3(M / 8), 256, 0, stream>>>(H, O, g1 + l * 256, be1 + l * 256, Hb);

        gemm256_kernel<true, true, 0><<<dim3(M / 128, 2), 256, 0, stream>>>(
            Hbu, (const unsigned int*)(W1T + (size_t)l * 65536), b1 + l * 256, nullptr, Qb);
        gemm256_kernel<true, false, 2><<<dim3(M / 128, 2), 256, 0, stream>>>(
            (const unsigned int*)Qb, (const unsigned int*)(W2T + (size_t)l * 65536),
            b2 + l * 256, O, nullptr);

        add_ln_kernel<<<dim3(M / 8), 256, 0, stream>>>(H, O, g2 + l * 256, be2 + l * 256, Hb);
    }

    head_kernel<<<dim3(M / 8), 256, 0, stream>>>(H, W_out, b_out, (float*)d_out);
}
